// HeteroEdgePromptPlus_64510408786220
// MI455X (gfx1250) — matrix operand order repaired and checked by reading, not compiled, not run on hardware
//
#include <hip/hip_runtime.h>

// ---------------------------------------------------------------------------
// Types for CDNA5 WMMA bf16 path
// ---------------------------------------------------------------------------
typedef __attribute__((ext_vector_type(16))) __bf16 bf16x16;
typedef __attribute__((ext_vector_type(8)))  __bf16 bf16x8;
typedef __attribute__((ext_vector_type(8)))  float  f32x8;
typedef __attribute__((__vector_size__(16))) int    v4i_t;   // matches builtin param

#define WMMA_BF16(a, b, c) \
    __builtin_amdgcn_wmma_f32_16x16x32_bf16(false, (a), false, (b), (short)0, (c), false, false)

// Async direct-to-LDS path (gfx1250), guarded so the file compiles on either
// toolchain; fallback is the VGPR round-trip.
#if defined(__has_builtin)
#if __has_builtin(__builtin_amdgcn_global_load_async_to_lds_b128) && \
    __has_builtin(__builtin_amdgcn_s_wait_asynccnt)
#define USE_ASYNC_LDS 1
#endif
#endif

static __device__ __forceinline__ bf16x16 cat16(bf16x8 lo, bf16x8 hi) {
    return __builtin_shufflevector(lo, hi, 0,1,2,3,4,5,6,7,8,9,10,11,12,13,14,15);
}

static __device__ __forceinline__ bf16x16 cvt16(float4 a, float4 b, float4 c, float4 d) {
    bf16x16 r;
    r[0]=(__bf16)a.x;  r[1]=(__bf16)a.y;  r[2]=(__bf16)a.z;  r[3]=(__bf16)a.w;
    r[4]=(__bf16)b.x;  r[5]=(__bf16)b.y;  r[6]=(__bf16)b.z;  r[7]=(__bf16)b.w;
    r[8]=(__bf16)c.x;  r[9]=(__bf16)c.y;  r[10]=(__bf16)c.z; r[11]=(__bf16)c.w;
    r[12]=(__bf16)d.x; r[13]=(__bf16)d.y; r[14]=(__bf16)d.z; r[15]=(__bf16)d.w;
    return r;
}

// Wave-internal LDS RAW fence: LDS pipe is in-order per wave; this stops the
// compiler from reordering the ds_loads above the ds_stores.
static __device__ __forceinline__ void wave_lds_fence() {
    __builtin_amdgcn_wave_barrier();
    asm volatile("s_wait_dscnt 0" ::: "memory");
    __builtin_amdgcn_wave_barrier();
}

// ---------------------------------------------------------------------------
// Kernel 0a: convert scorer weights Ws [16,256] -> bf16 row-major, and build
// AT [128,32] = A^T zero-padded to K=32 (bf16), for the output-stage WMMA B.
// ---------------------------------------------------------------------------
__global__ void prep_weights_kernel(const float* __restrict__ Ws,
                                    const float* __restrict__ A,
                                    __bf16* __restrict__ WsB,
                                    __bf16* __restrict__ ATB) {
    int tid = threadIdx.x;
    for (int i = tid; i < 16 * 256; i += 256) WsB[i] = (__bf16)Ws[i];
    for (int i = tid; i < 128 * 32; i += 256) {
        int col = i >> 5, k = i & 31;
        ATB[i] = (k < 16) ? (__bf16)A[k * 128 + col] : (__bf16)0.0f;
    }
}

// Kernel 0b: bulk f32 -> bf16 conversion (for Wp matrices; loop-invariant
// across all projection tiles, so convert once instead of per WMMA B-frag).
__global__ void prep_bf16_kernel(const float* __restrict__ src,
                                 __bf16* __restrict__ dst, int n) {
    int i = blockIdx.x * blockDim.x + threadIdx.x;
    int stride = gridDim.x * blockDim.x;
    for (; i < n; i += stride) dst[i] = (__bf16)src[i];
}

// ---------------------------------------------------------------------------
// Kernel 1: projection  xp[N,128] = bf16( x[N,D] @ Wp[128,D]^T + bp )
// One wave computes a 16-row x 128-col tile (8 WMMA accumulators, K-outer).
// Wp is pre-converted bf16; x is converted on the fly (read exactly once).
// ---------------------------------------------------------------------------
__global__ void __launch_bounds__(128)
project_kernel(const float* __restrict__ x, const __bf16* __restrict__ Wpb,
               const float* __restrict__ bp, __bf16* __restrict__ xp,
               int N, int D) {
    const int lane    = threadIdx.x & 31;
    const int wave    = threadIdx.x >> 5;
    const int halfsel = lane >> 4;   // 0: lanes 0-15, 1: lanes 16-31
    const int l16     = lane & 15;
    const int gwave   = blockIdx.x * (blockDim.x >> 5) + wave;
    const int nwaves  = gridDim.x * (blockDim.x >> 5);
    const int ntiles  = (N + 15) >> 4;

    for (int tile = gwave; tile < ntiles; tile += nwaves) {
        const int r0  = tile << 4;
        int row = r0 + l16; if (row >= N) row = N - 1;
        const float* xr = x + (size_t)row * D;

        f32x8 acc[8];
        #pragma unroll
        for (int cc = 0; cc < 8; ++cc) acc[cc] = (f32x8)0.0f;

        for (int kb = 0; kb < D; kb += 32) {
            // A fragment: row = l16; K = kb + {0..7,16..23} (+8 for upper half-wave)
            const int s = halfsel * 8;
            float4 f0 = *(const float4*)&xr[kb + s];
            float4 f1 = *(const float4*)&xr[kb + s + 4];
            float4 f2 = *(const float4*)&xr[kb + s + 16];
            float4 f3 = *(const float4*)&xr[kb + s + 20];
            bf16x16 afrag = cvt16(f0, f1, f2, f3);

            #pragma unroll
            for (int cc = 0; cc < 8; ++cc) {
                // B fragment: col = cc*16+l16; K = kb + {0..7,16..23} (+8 for upper half-wave)
                const int col = cc * 16 + l16;
                const __bf16* wr = Wpb + (size_t)col * D + kb + halfsel * 8;
                bf16x8 g0 = *(const bf16x8*)&wr[0];
                bf16x8 g1 = *(const bf16x8*)&wr[16];
                bf16x16 bfrag = cat16(g0, g1);
                acc[cc] = WMMA_BF16(afrag, bfrag, acc[cc]);
            }
        }

        #pragma unroll
        for (int cc = 0; cc < 8; ++cc) {
            const int col = cc * 16 + l16;
            const float bias = bp[col];
            #pragma unroll
            for (int r = 0; r < 8; ++r) {
                const int rr = r0 + r + halfsel * 8;
                if (rr < N) xp[(size_t)rr * 128 + col] = (__bf16)(acc[cc][r] + bias);
            }
        }
    }
}

// ---------------------------------------------------------------------------
// Kernel 2: edge prompt.  Per 16-edge tile (one wave):
//   feat[16][256] = concat(xp_src[ei0], xp_dst[ei1])   (async global->LDS)
//   logits[16,16] = feat @ Ws^T        (8x v_wmma_f32_16x16x32_bf16)
//   b = softmax(leaky_relu(logits + bs))  (VALU + 16-lane shfl_xor butterflies)
//   out[16,128]   = b @ A              (8x WMMA, b re-laid-out through LDS)
// ---------------------------------------------------------------------------
__global__ void __launch_bounds__(128)
edge_prompt_kernel(const __bf16* __restrict__ xp_src,
                   const __bf16* __restrict__ xp_dst,
                   const int* __restrict__ ei, int E,
                   const __bf16* __restrict__ WsB,   // [16][256] bf16
                   const __bf16* __restrict__ ATB,   // [128][32] bf16 (A^T, K-padded)
                   const float* __restrict__ bs,
                   float* __restrict__ out) {
    __shared__ __bf16 sWs[16 * 256];          // 8 KB
    __shared__ __bf16 sAT[128 * 32];          // 8 KB
    __shared__ __bf16 sFeat[4][16 * 256];     // 32 KB (per-wave)
    __shared__ __bf16 sB[4][16 * 32];         // 4 KB  (per-wave)

    const int tid = threadIdx.x;
    // Block-cooperative constant load into LDS (Ws: 2048 dwords, AT: 2048 dwords)
    {
        const uint32_t* gw = (const uint32_t*)WsB;
        uint32_t*       lw = (uint32_t*)sWs;
        for (int i = tid; i < 2048; i += 128) lw[i] = gw[i];
        const uint32_t* ga = (const uint32_t*)ATB;
        uint32_t*       la = (uint32_t*)sAT;
        for (int i = tid; i < 2048; i += 128) la[i] = ga[i];
    }
    __syncthreads();

    const int lane    = tid & 31;
    const int wave    = tid >> 5;
    const int halfsel = lane >> 4;
    const int l16     = lane & 15;
    __bf16* feat = sFeat[wave];
    __bf16* bbuf = sB[wave];

    // zero the K=16..31 pad region of the probability tile (written once)
    {
        int4 z = {0, 0, 0, 0};
        *(int4*)&bbuf[l16 * 32 + 16 + halfsel * 8] = z;
    }

    const float bsv = bs[l16];
    const int ntiles = (E + 15) >> 4;
    const int gwave  = blockIdx.x * 4 + wave;
    const int nwaves = gridDim.x * 4;

    for (int tile = gwave; tile < ntiles; tile += nwaves) {
        const int e0 = tile << 4;

        // ---- stage 16 edges x (u||v) into LDS: lane -> (edge = lane/2, half)
        {
            const int edge = lane >> 1;
            const int half = lane & 1;          // 0: src endpoint, 1: dst endpoint
            int e = e0 + edge; if (e >= E) e = E - 1;
            const int idx = ei[half * E + e];
            const __bf16* rowp = (half ? xp_dst : xp_src) + (size_t)idx * 128;
            __bf16* lrow = &feat[edge * 256 + half * 128];
#ifdef USE_ASYNC_LDS
            // 128 bf16 = 256 B per lane, as 16 async b128 transfers that
            // bypass VGPRs entirely (tracked by ASYNCcnt).  Offsets must be
            // frontend-constant, so expand with literal immediates.
            auto gp = (__attribute__((address_space(1))) v4i_t*)rowp;
            auto lp = (__attribute__((address_space(3))) v4i_t*)lrow;
#define ASYNC_CP(c) __builtin_amdgcn_global_load_async_to_lds_b128(gp, lp, (c) * 16, 0)
            ASYNC_CP(0);  ASYNC_CP(1);  ASYNC_CP(2);  ASYNC_CP(3);
            ASYNC_CP(4);  ASYNC_CP(5);  ASYNC_CP(6);  ASYNC_CP(7);
            ASYNC_CP(8);  ASYNC_CP(9);  ASYNC_CP(10); ASYNC_CP(11);
            ASYNC_CP(12); ASYNC_CP(13); ASYNC_CP(14); ASYNC_CP(15);
#undef ASYNC_CP
#else
            const int4* g = (const int4*)rowp;
            int4* l = (int4*)lrow;
            #pragma unroll
            for (int c = 0; c < 16; ++c) l[c] = g[c];
#endif
        }
#ifdef USE_ASYNC_LDS
        __builtin_amdgcn_wave_barrier();
        __builtin_amdgcn_s_wait_asynccnt(0);
        __builtin_amdgcn_wave_barrier();
#else
        wave_lds_fence();
#endif

        // ---- logits: [16 edges] x [K=256] x [16 scorers]
        f32x8 acc = (f32x8)0.0f;
        #pragma unroll
        for (int kb = 0; kb < 256; kb += 32) {
            const int s = halfsel * 8;
            bf16x8 alo = *(const bf16x8*)&feat[l16 * 256 + kb + s];
            bf16x8 ahi = *(const bf16x8*)&feat[l16 * 256 + kb + s + 16];
            bf16x16 afrag = cat16(alo, ahi);
            const int koff = halfsel * 8;
            bf16x8 blo = *(const bf16x8*)&sWs[l16 * 256 + kb + koff];
            bf16x8 bhi = *(const bf16x8*)&sWs[l16 * 256 + kb + koff + 16];
            bf16x16 bfrag = cat16(blo, bhi);
            acc = WMMA_BF16(afrag, bfrag, acc);
        }

        // ---- bias + leaky_relu + softmax along N (= lane dim, 16-lane groups)
        float prob[8];
        #pragma unroll
        for (int r = 0; r < 8; ++r) {
            float lg = acc[r] + bsv;
            lg = lg > 0.0f ? lg : 0.01f * lg;
            float mx = lg;
            mx = fmaxf(mx, __shfl_xor(mx, 1, 32));
            mx = fmaxf(mx, __shfl_xor(mx, 2, 32));
            mx = fmaxf(mx, __shfl_xor(mx, 4, 32));
            mx = fmaxf(mx, __shfl_xor(mx, 8, 32));
            float ex = __expf(lg - mx);
            float sm = ex;
            sm += __shfl_xor(sm, 1, 32);
            sm += __shfl_xor(sm, 2, 32);
            sm += __shfl_xor(sm, 4, 32);
            sm += __shfl_xor(sm, 8, 32);
            prob[r] = ex / sm;
        }

        // ---- re-layout probability tile through LDS into WMMA-A form
        #pragma unroll
        for (int r = 0; r < 8; ++r) {
            const int m = r + halfsel * 8;          // edge row
            bbuf[m * 32 + l16] = (__bf16)prob[r];   // k = l16 (k>=16 stays 0)
        }
        wave_lds_fence();

        bf16x8 balo = *(const bf16x8*)&bbuf[l16 * 32 + halfsel * 8];
        bf16x8 bahi = *(const bf16x8*)&bbuf[l16 * 32 + halfsel * 8 + 16];  // zeros
        bf16x16 bA = cat16(balo, bahi);

        // ---- out[16,128] = b @ A, 8 column tiles
        #pragma unroll
        for (int cc = 0; cc < 8; ++cc) {
            const int col = cc * 16 + l16;
            bf16x8 wlo = *(const bf16x8*)&sAT[col * 32 + halfsel * 8];
            bf16x8 whi = *(const bf16x8*)&sAT[col * 32 + halfsel * 8 + 16];
            bf16x16 bW = cat16(wlo, whi);
            f32x8 o = WMMA_BF16(bA, bW, (f32x8)0.0f);
            #pragma unroll
            for (int r = 0; r < 8; ++r) {
                const int e = e0 + r + halfsel * 8;
                if (e < E) out[(size_t)e * 128 + col] = o[r];  // coalesced rows
            }
        }
    }
}

// ---------------------------------------------------------------------------
// Host launcher
// ---------------------------------------------------------------------------
extern "C" void kernel_launch(void* const* d_in, const int* in_sizes, int n_in,
                              void* d_out, int out_size, void* d_ws, size_t ws_size,
                              hipStream_t stream) {
    const float* x_paper   = (const float*)d_in[0];
    const float* x_author  = (const float*)d_in[1];
    const float* Wp_paper  = (const float*)d_in[2];
    const float* bp_paper  = (const float*)d_in[3];
    const float* Wp_author = (const float*)d_in[4];
    const float* bp_author = (const float*)d_in[5];
    const float* Ws_writes = (const float*)d_in[6];
    const float* bs_writes = (const float*)d_in[7];
    const float* A_writes  = (const float*)d_in[8];
    const float* Ws_cites  = (const float*)d_in[9];
    const float* bs_cites  = (const float*)d_in[10];
    const float* A_cites   = (const float*)d_in[11];
    const int*   ei_writes = (const int*)d_in[12];
    const int*   ei_cites  = (const int*)d_in[13];

    const int Dp = 256, Da = 128;
    const int Np = in_sizes[0] / Dp;
    const int Na = in_sizes[1] / Da;
    const int E  = in_sizes[12] / 2;

    // workspace layout (256-byte aligned)
    char* w = (char*)d_ws;
    size_t off = 0;
    auto al = [](size_t v) { return (v + 255) & ~(size_t)255; };
    __bf16* xpP  = (__bf16*)(w + off); off += al((size_t)Np * 128 * 2);
    __bf16* xpA  = (__bf16*)(w + off); off += al((size_t)Na * 128 * 2);
    __bf16* WsW  = (__bf16*)(w + off); off += 16 * 256 * 2;
    __bf16* ATW  = (__bf16*)(w + off); off += 128 * 32 * 2;
    __bf16* WsC  = (__bf16*)(w + off); off += 16 * 256 * 2;
    __bf16* ATC  = (__bf16*)(w + off); off += 128 * 32 * 2;
    __bf16* WpPb = (__bf16*)(w + off); off += al((size_t)128 * Dp * 2);
    __bf16* WpAb = (__bf16*)(w + off); off += al((size_t)128 * Da * 2);

    // 0) weight prep (bf16 + transpose/pad)
    prep_weights_kernel<<<1, 256, 0, stream>>>(Ws_writes, A_writes, WsW, ATW);
    prep_weights_kernel<<<1, 256, 0, stream>>>(Ws_cites,  A_cites,  WsC, ATC);
    prep_bf16_kernel<<<32, 256, 0, stream>>>(Wp_paper,  WpPb, 128 * Dp);
    prep_bf16_kernel<<<16, 256, 0, stream>>>(Wp_author, WpAb, 128 * Da);

    // 1) projections to bf16 prompt embeddings (L2-resident gather tables)
    {
        int tilesP = (Np + 15) / 16;
        int tilesA = (Na + 15) / 16;
        int blkP = (tilesP + 3) / 4;
        int blkA = (tilesA + 3) / 4;
        project_kernel<<<blkP, 128, 0, stream>>>(x_paper,  WpPb, bp_paper,  xpP, Np, Dp);
        project_kernel<<<blkA, 128, 0, stream>>>(x_author, WpAb, bp_author, xpA, Na, Da);
    }

    // 2) edge prompts (two edge types)
    {
        int tiles  = (E + 15) / 16;
        int blocks = (tiles + 3) / 4;
        if (blocks > 4096) blocks = 4096;
        edge_prompt_kernel<<<blocks, 128, 0, stream>>>(
            xpA, xpP, ei_writes, E, WsW, ATW, bs_writes, (float*)d_out);
        edge_prompt_kernel<<<blocks, 128, 0, stream>>>(
            xpP, xpP, ei_cites, E, WsC, ATC, bs_cites, (float*)d_out + (size_t)E * 128);
    }
}
